// MsaHmmCell_82446192214255
// MI455X (gfx1250) — matrix operand order repaired and checked by reading, not compiled, not run on hardware
//
#include <hip/hip_runtime.h>
#include <hip/hip_bf16.h>
#include <math.h>

// ---------------------------------------------------------------------------
// Profile-HMM forward (MsaHmmCell) for MI455X / gfx1250.
//
// Batch rows are independent recurrences: each workgroup owns 16 rows and
// runs the full 1024-step scan with alpha resident in LDS (bf16). Per step,
// new_alpha[16 x 1056] = alpha[16 x 1056] @ A[1056 x 1056] via
// V_WMMA_F32_16X16X32_BF16, then emission gather, LDS-atomic row sums,
// renormalize, log-accumulate.
//
// Sparsity: A's nonzero pattern by column n is
//   n in [1,512]   : k in {0} U [1,n-1] U {511+n} U {1024}   (banded match)
//   n in [513,1023]: k in {n-512, n}                          (2 K-tiles!)
//   n in {1024..6} : k in [0,512] U {1024}
// Each 16-column tile needs a contiguous K-tile prefix plus <=3 extras,
// computed analytically. The wave id is taken through readfirstlane so the
// whole schedule is SGPR-resident -> uniform scalar branches, full-EXEC
// WMMA, and a pipelineable counted inner loop.
// ---------------------------------------------------------------------------

#define LM     512
#define Q      1027          // 2*LM + 3 real states
#define QP     1056          // padded to 66*16 = 33*32
#define NT     66            // N tiles of 16
#define KT     33            // K tiles of 32
#define TSTEPS 1024
#define ALDS   1064          // LDS row stride (ushorts), 16B aligned rows
#define TINYF  1.17549435e-38f

typedef __attribute__((ext_vector_type(16))) __bf16 v16bf;
typedef __attribute__((ext_vector_type(8)))  float  v8f;

union Frag16 { v16bf v; uint4 q[2]; unsigned short u[16]; };

__device__ __forceinline__ unsigned short f2bf(float f) {
  unsigned int u = __float_as_uint(f);
  u += 0x7FFFu + ((u >> 16) & 1u);           // round-to-nearest-even
  return (unsigned short)(u >> 16);
}
__device__ __forceinline__ float bf2f(unsigned short b) {
  return __uint_as_float(((unsigned int)b) << 16);
}

// ---- workspace float layout (after Atr and BmT and init) -------------------
// par +    0 : btm_p[512]       par + 2560 : i2m_p[512]    par + 5120 : enter[512]
// par +  512 : mtm_p[512]       par + 3072 : i2i_p[512]    par + 5632 : m2e_tot[512]
// par + 1024 : m2i_p[512]       par + 3584 : dtm_p[512]    par + 6144 : scal[16]
// par + 1536 : (unused)         par + 4096 : dtd_p[512]
// par + 2048 : mtd_p[512]       par + 4608 : cp_log[512]
// scal: 0 floop, 1 fexit, 2 ep0, 3 ep1, 4 ep2, 5 begin_end, 6 p0, 7 begin_p[L]

__global__ __launch_bounds__(512)
void build_params_kernel(const float* __restrict__ btm, const float* __restrict__ m2e,
                         const float* __restrict__ mtm, const float* __restrict__ m2i,
                         const float* __restrict__ i2m, const float* __restrict__ i2i,
                         const float* __restrict__ mtd, const float* __restrict__ dtm,
                         const float* __restrict__ dtd, const float* __restrict__ lfl,
                         const float* __restrict__ lfe, const float* __restrict__ e2u,
                         const float* __restrict__ e2r, const float* __restrict__ e2t,
                         const float* __restrict__ flank,
                         float* __restrict__ par, float* __restrict__ init) {
  __shared__ float red[512];
  const int tid = threadIdx.x;
  float* btm_p   = par;
  float* mtm_p   = par +  512;
  float* m2i_p   = par + 1024;
  float* mtd_p   = par + 2048;
  float* i2m_p   = par + 2560;
  float* i2i_p   = par + 3072;
  float* dtm_p   = par + 3584;
  float* dtd_p   = par + 4096;
  float* cp_log  = par + 4608;
  float* enter_  = par + 5120;
  float* m2e_tot = par + 5632;
  float* scal    = par + 6144;

  // ---- begin_p = softmax(concat(btm[512], mtd[0])) : 513 elements ----------
  float v = btm[tid];
  float extra = mtd[0];
  red[tid] = (tid == 0) ? fmaxf(v, extra) : v;
  __syncthreads();
  for (int s = 256; s > 0; s >>= 1) { if (tid < s) red[tid] = fmaxf(red[tid], red[tid + s]); __syncthreads(); }
  float mx = red[0]; __syncthreads();
  float ev = expf(v - mx);
  red[tid] = ev + ((tid == 0) ? expf(extra - mx) : 0.f);
  __syncthreads();
  for (int s = 256; s > 0; s >>= 1) { if (tid < s) red[tid] += red[tid + s]; __syncthreads(); }
  float den = red[0]; __syncthreads();
  btm_p[tid] = ev / den;
  if (tid == 0) { float bpl = expf(extra - mx) / den; scal[7] = bpl; mtd_p[0] = bpl; }

  float m2e_p_loc = 0.f;   // register-resident; only needed for m2e_tot below
  if (tid < LM - 1) {
    // 4-way softmax over (mtm, m2i, m2e[:L-1], mtd[1:])
    float a = mtm[tid], b = m2i[tid], c = m2e[tid], d = mtd[tid + 1];
    float m4 = fmaxf(fmaxf(a, b), fmaxf(c, d));
    float ea = expf(a - m4), eb = expf(b - m4), ec = expf(c - m4), ed = expf(d - m4);
    float s4 = ea + eb + ec + ed;
    mtm_p[tid] = ea / s4; m2i_p[tid] = eb / s4; m2e_p_loc = ec / s4; mtd_p[tid + 1] = ed / s4;
    // 2-way insert softmax
    float ia = i2m[tid], ib = i2i[tid];
    float im = fmaxf(ia, ib); float eia = expf(ia - im), eib = expf(ib - im);
    float is = eia + eib;
    i2m_p[tid] = eia / is; i2i_p[tid] = eib / is;
    // 2-way delete softmax
    float da = dtm[tid], db = dtd[tid];
    float dm = fmaxf(da, db); float eda = expf(da - dm), edb = expf(db - dm);
    float dsm = eda + edb;
    dtm_p[tid] = eda / dsm; dtd_p[tid] = edb / dsm;
  }
  if (tid == 0) {
    dtm_p[LM - 1] = 1.f;
    float a = lfl[0], b = lfe[0]; float m2 = fmaxf(a, b);
    float ea = expf(a - m2), eb = expf(b - m2); float s2 = ea + eb;
    scal[0] = ea / s2; scal[1] = eb / s2;                       // floop, fexit
    float u = e2u[0], r = e2r[0], t = e2t[0];
    float m3 = fmaxf(fmaxf(u, r), t);
    float eu = expf(u - m3), er = expf(r - m3), et = expf(t - m3);
    float s3 = eu + er + et;
    scal[2] = eu / s3; scal[3] = er / s3; scal[4] = et / s3;    // ep0..2
    scal[6] = 1.f / (1.f + expf(-flank[0]));                    // p0
  }
  __syncthreads();
  if (tid == 0) {            // cumulative log of delete chain
    cp_log[0] = 0.f;
    float acc = 0.f;
    for (int i = 0; i < LM - 1; ++i) { acc += logf(dtd_p[i]); cp_log[i + 1] = acc; }
  }
  __syncthreads();
  {
    // Sk[i][j] = mtd_p[i] * exp(cp_log[j]-cp_log[i]) * dtm_p[j]   (j >= i)
    float mtd0 = mtd_p[0];
    int j = tid;
    enter_[j] = btm_p[j] + ((j >= 1) ? (mtd0 * expf(cp_log[j - 1]) * dtm_p[j - 1]) : 0.f);
    if (tid == 0) scal[5] = mtd0 * expf(cp_log[LM - 1]) * dtm_p[LM - 1];   // begin_end
    if (tid < LM - 1) {
      float ski = mtd_p[tid + 1] * expf(cp_log[LM - 1] - cp_log[tid + 1]) * dtm_p[LM - 1];
      m2e_tot[tid] = m2e_p_loc + ski;
    }
    if (tid == 0) m2e_tot[LM - 1] = 1.f;
  }
  __syncthreads();
  {
    float p0 = scal[6], be = scal[5];
    for (int n = tid; n < QP; n += 512) {
      float x = 0.f;
      if      (n == 0)          x = p0;
      else if (n <= LM)         x = (1.f - p0) * enter_[n - 1];
      else if (n == 2 * LM)     x = (1.f - p0) * be * scal[2];
      else if (n == 2 * LM + 1) x = (1.f - p0) * be * scal[3];
      else if (n == 2 * LM + 2) x = (1.f - p0) * be * scal[4];
      init[n] = x;
    }
  }
}

// Fill transposed transition matrix Atr[n][k] = A[k][n] in bf16.
__global__ __launch_bounds__(256)
void build_A_kernel(const float* __restrict__ par, unsigned short* __restrict__ Atr) {
  long long idx = (long long)blockIdx.x * 256 + threadIdx.x;
  if (idx >= (long long)QP * QP) return;
  int k = (int)(idx / QP);
  int n = (int)(idx - (long long)k * QP);
  const float* mtm_p   = par +  512;
  const float* m2i_p   = par + 1024;
  const float* mtd_p   = par + 2048;
  const float* i2m_p   = par + 2560;
  const float* i2i_p   = par + 3072;
  const float* dtm_p   = par + 3584;
  const float* cp_log  = par + 4608;
  const float* enter_  = par + 5120;
  const float* m2e_tot = par + 5632;
  const float* scal    = par + 6144;
  const float floop = scal[0], fexit = scal[1];
  const float ep0 = scal[2], ep1 = scal[3], ep2 = scal[4], be = scal[5];
  const int U = 2 * LM, RF = 2 * LM + 1, TM = 2 * LM + 2;
  float a = 0.f;
  if (k < Q && n < Q) {
    if (k == 0) {                                   // unaligned-flank state
      if      (n == 0)  a = floop;
      else if (n <= LM) a = fexit * enter_[n - 1];
      else if (n == U)  a = fexit * be * ep0;
      else if (n == RF) a = fexit * be * ep1;
      else if (n == TM) a = fexit * be * ep2;
    } else if (k <= LM) {                           // match rows
      int i = k - 1;
      if (n >= 1 && n <= LM) {
        int j = n - 1;
        if (j == i + 1) a += mtm_p[i];
        if (i <= LM - 2 && j - 1 >= i + 1)          // Sk[i+1][j-1] (delete skips)
          a += mtd_p[i + 1] * expf(cp_log[j - 1] - cp_log[i + 1]) * dtm_p[j - 1];
      } else if (k <= LM - 1 && n == LM + k) {
        a = m2i_p[k - 1];                           // match -> insert
      } else if (n == U)  a = m2e_tot[i] * ep0;
      else if (n == RF)   a = m2e_tot[i] * ep1;
      else if (n == TM)   a = m2e_tot[i] * ep2;
    } else if (k <= 2 * LM - 1) {                   // insert rows
      int i = k - (LM + 1);
      if      (n == 2 + i) a = i2m_p[i];
      else if (n == k)     a = i2i_p[i];
    } else if (k == U) {
      if      (n >= 1 && n <= LM) a = fexit * enter_[n - 1];
      else if (n == U)  a = floop + fexit * be * ep0;
      else if (n == RF) a = fexit * be * ep1;
      else if (n == TM) a = fexit * be * ep2;
    } else if (k == RF) {
      if      (n == RF) a = floop;
      else if (n == TM) a = fexit;
    } else {                                        // k == TM
      if (n == TM) a = 1.f;
    }
  }
  Atr[(long long)n * QP + k] = f2bf(a);
}

// BmT[token][state] (32 padded tokens x QP padded states), f32.
__global__ __launch_bounds__(256)
void build_emis_kernel(const float* __restrict__ em, const float* __restrict__ ins,
                       float* __restrict__ BmT) {
  int s = blockIdx.x * 256 + threadIdx.x;
  if (s >= QP) return;
  float row[25];
  bool valid = false;
  if (s == 0 || (s >= LM + 1 && s <= 2 * LM + 1)) {
    for (int t = 0; t < 25; ++t) row[t] = ins[t];
    valid = true;
  } else if (s >= 1 && s <= LM) {
    for (int t = 0; t < 25; ++t) row[t] = em[(s - 1) * 25 + t];
    valid = true;
  }
  if (valid) {
    float m = row[0];
    for (int t = 1; t < 25; ++t) m = fmaxf(m, row[t]);
    float sum = 0.f;
    for (int t = 0; t < 25; ++t) { row[t] = expf(row[t] - m); sum += row[t]; }
    float inv = 1.f / sum;
    for (int t = 0;  t < 25; ++t) BmT[(size_t)t * QP + s] = row[t] * inv;
    for (int t = 25; t < 32; ++t) BmT[(size_t)t * QP + s] = 0.f;
  } else {
    for (int t = 0; t < 32; ++t)
      BmT[(size_t)t * QP + s] = (t == 25 && s == 2 * LM + 2) ? 1.f : 0.f;
  }
}

// ---------------------------------------------------------------------------
// Forward scan: one workgroup = 16 batch rows, 16 waves, alpha lives in LDS.
// ---------------------------------------------------------------------------
__global__ __launch_bounds__(512)
void hmm_forward_kernel(const int* __restrict__ seq,
                        const unsigned short* __restrict__ Atr,
                        const float* __restrict__ BmT,
                        const float* __restrict__ init,
                        float* __restrict__ ll_out) {
  const int b0   = blockIdx.x * 16;
  const int tid  = threadIdx.x;
  const int lane = tid & 31;
  // readfirstlane: wave id is uniform; putting it in an SGPR makes the whole
  // tile/K schedule scalar (uniform branches, full-EXEC WMMA, pipelineable).
  const int w    = __builtin_amdgcn_readfirstlane(tid >> 5);

  __shared__ unsigned short s_alpha[16 * ALDS];   // bf16 alpha, 16 rows
  __shared__ float s_rowsum[2][16];
  __shared__ float s_ll[16];
  __shared__ __align__(16) int s_rowoff[16];      // tok[m] * QP (precomputed)

  if (tid < 16) {
    s_rowsum[0][tid] = 0.f; s_rowsum[1][tid] = 0.f; s_ll[tid] = 0.f;
    s_rowoff[tid] = seq[(size_t)(b0 + tid) * TSTEPS + 0] * QP;
  }
  __syncthreads();

  // ---- t = 0: alpha = init * E0, rowsum, ll, normalize ---------------------
  for (int idx = tid; idx < 16 * QP; idx += 512) {
    int m = idx / QP, n = idx - m * QP;
    float x = init[n] * BmT[s_rowoff[m] + n];
    s_alpha[m * ALDS + n] = f2bf(x);
    atomicAdd(&s_rowsum[0][m], x);
  }
  __syncthreads();
  if (tid < 16) {
    s_ll[tid] = logf(s_rowsum[0][tid] + TINYF);
    s_rowoff[tid] = seq[(size_t)(b0 + tid) * TSTEPS + 1] * QP;
  }
  __syncthreads();
  for (int idx = tid; idx < 16 * QP; idx += 512) {
    int m = idx / QP, n = idx - m * QP;
    float s = s_rowsum[0][m] + TINYF;
    s_alpha[m * ALDS + n] = f2bf(bf2f(s_alpha[m * ALDS + n]) / s);
  }
  __syncthreads();

  // ---- main scan -----------------------------------------------------------
  const int m16   = lane & 15;
  const int kaoff = (lane & 16) ? 16 : 0;    // A-operand K sub-offset
  const int kboff = (lane & 16) ? 16 : 0;    // B-operand K sub-offset
  const int moff  = (lane & 16) ? 8 : 0;     // C/D row offset

  for (int t = 1; t < TSTEPS; ++t) {
    const int cur = t & 1, nxt = cur ^ 1;
    float keep[5][8];
    float rloc[8];                           // per-lane row-sum contributions

    // Emission row offsets for this step: two ds_load_b128 per lane.
    int ro[8];
    {
      const int4* rp = (const int4*)&s_rowoff[moff];
      int4 ra = rp[0], rb = rp[1];
      ro[0] = ra.x; ro[1] = ra.y; ro[2] = ra.z; ro[3] = ra.w;
      ro[4] = rb.x; ro[5] = rb.y; ro[6] = rb.z; ro[7] = rb.w;
    }
#pragma unroll
    for (int v = 0; v < 8; ++v) rloc[v] = 0.f;

    // ---- phase 1: sparse-scheduled tile GEMMs + emission multiply ----------
#pragma unroll
    for (int slot = 0; slot < 5; ++slot) {
      int tile = w + slot * 16;              // SGPR (w is readfirstlane'd)
      if (tile >= NT) break;
      v8f acc = {0.f, 0.f, 0.f, 0.f, 0.f, 0.f, 0.f, 0.f};
      const int ng = tile * 16 + m16;
      const unsigned short* bcol = Atr + (size_t)ng * QP + kboff;

      auto kbody = [&](int kt) {
        Frag16 af, bf;
        const unsigned short* ap = &s_alpha[m16 * ALDS + kt * 32 + kaoff];
        af.q[0] = *(const uint4*)(ap);        // K = kaoff + 0..7
        af.q[1] = *(const uint4*)(ap + 8);    // K = kaoff + 8..15
        const unsigned short* bp = bcol + (size_t)kt * 32;
        bf.q[0] = *(const uint4*)(bp);        // K = kboff + 0..7
        bf.q[1] = *(const uint4*)(bp + 8);    // K = kboff + 8..15
        acc = __builtin_amdgcn_wmma_f32_16x16x32_bf16(
                  false, af.v, false, bf.v, (short)0, acc, false, false);
      };

      // Analytic K-tile schedule (all-scalar): prefix [0..ktmax] + <=3 extras.
      int ktmax, e0 = -1, e1 = -1, e2 = -1;
      if (tile <= 32) {                       // match-region columns
        ktmax = tile >> 1;                    // covers k=0 + triangular band
        int a = (511 + 16 * tile) >> 5;       // insert->match band (i2m)
        int b = (526 + 16 * tile) >> 5;
        e0 = a;
        e1 = (b > a) ? b : -1;
        int last = (b > a) ? b : a;
        e2 = (last < 32) ? 32 : -1;           // U row (k=1024)
      } else if (tile <= 63) {                // pure insert columns: 2 K-tiles
        ktmax = -1;
        e0 = (tile >> 1) - 16;                // match->insert diagonal (m2i)
        e1 = tile >> 1;                       // insert self-loop (i2i)
      } else if (tile == 64) {                // U/RF/TM columns
        ktmax = 16;                           // k in [0,512]
        e0 = 32;                              // k in {1024..1026}
      } else {                                // tile 65: all padding
        ktmax = -1;
      }
#pragma unroll 2
      for (int kt = 0; kt <= ktmax; ++kt) kbody(kt);
      if (e0 >= 0) kbody(e0);
      if (e1 >= 0) kbody(e1);
      if (e2 >= 0) kbody(e2);

      // Emission multiply; row sums accumulate in registers across tiles
      // (row m depends only on lane, not tile).
      float evv[8];
#pragma unroll
      for (int v = 0; v < 8; ++v) evv[v] = BmT[ro[v] + ng];
#pragma unroll
      for (int v = 0; v < 8; ++v) {
        float x = acc[v] * evv[v];
        keep[slot][v] = x;
        rloc[v] += x;
      }
    }
    // One batch of 8 LDS atomics per lane per step (was 8 per tile).
    {
      float* rs = &s_rowsum[cur][moff];
#pragma unroll
      for (int v = 0; v < 8; ++v) atomicAdd(&rs[v], rloc[v]);   // ds_add_f32
    }
    __syncthreads();

    // ---- phase 2: normalize + write back bf16 alpha ------------------------
#pragma unroll
    for (int slot = 0; slot < 5; ++slot) {
      int tile = w + slot * 16;
      if (tile >= NT) break;
      const int ng = tile * 16 + m16;
#pragma unroll
      for (int v = 0; v < 8; ++v) {
        float s = s_rowsum[cur][v + moff] + TINYF;
        s_alpha[(v + moff) * ALDS + ng] = f2bf(keep[slot][v] / s);
      }
    }
    if (tid < 16) {
      float s = s_rowsum[cur][tid] + TINYF;
      s_ll[tid] += logf(s);
      s_rowsum[nxt][tid] = 0.f;
      if (t + 1 < TSTEPS) s_rowoff[tid] = seq[(size_t)(b0 + tid) * TSTEPS + (t + 1)] * QP;
    }
    __syncthreads();
  }

  if (tid < 16) ll_out[b0 + tid] = s_ll[tid];
}

// ---------------------------------------------------------------------------
extern "C" void kernel_launch(void* const* d_in, const int* in_sizes, int n_in,
                              void* d_out, int out_size, void* d_ws, size_t ws_size,
                              hipStream_t stream) {
  (void)in_sizes; (void)n_in; (void)out_size; (void)ws_size;
  const int*   seq   = (const int*)  d_in[0];
  const float* em    = (const float*)d_in[1];
  const float* ins   = (const float*)d_in[2];
  const float* flank = (const float*)d_in[3];
  const float* btm   = (const float*)d_in[4];
  const float* m2e   = (const float*)d_in[5];
  const float* mtm   = (const float*)d_in[6];
  const float* m2i   = (const float*)d_in[7];
  const float* i2m   = (const float*)d_in[8];
  const float* i2i   = (const float*)d_in[9];
  const float* mtd   = (const float*)d_in[10];
  const float* dtm   = (const float*)d_in[11];
  const float* dtd   = (const float*)d_in[12];
  const float* lfl   = (const float*)d_in[13];
  const float* lfe   = (const float*)d_in[14];
  const float* e2u   = (const float*)d_in[15];
  const float* e2r   = (const float*)d_in[16];
  const float* e2t   = (const float*)d_in[17];

  char* ws = (char*)d_ws;
  unsigned short* Atr  = (unsigned short*)ws;                               // QP*QP bf16
  float*          BmT  = (float*)(ws + (size_t)QP * QP * 2);                // 32*QP f32
  float*          initv= (float*)(ws + (size_t)QP * QP * 2 + (size_t)32 * QP * 4);
  float*          par  = (float*)(ws + (size_t)QP * QP * 2 + (size_t)32 * QP * 4
                                     + (size_t)QP * 4);                     // 6160 f32

  build_params_kernel<<<1, 512, 0, stream>>>(btm, m2e, mtm, m2i, i2m, i2i, mtd,
                                             dtm, dtd, lfl, lfe, e2u, e2r, e2t,
                                             flank, par, initv);
  build_emis_kernel<<<(QP + 255) / 256, 256, 0, stream>>>(em, ins, BmT);
  {
    long long total = (long long)QP * QP;
    build_A_kernel<<<(unsigned)((total + 255) / 256), 256, 0, stream>>>(par, Atr);
  }
  hmm_forward_kernel<<<8, 512, 0, stream>>>(seq, Atr, BmT, initv, (float*)d_out);
}
